// CrossAttention_11836929868369
// MI455X (gfx1250) — matrix operand order repaired and checked by reading, not compiled, not run on hardware
//
#include <hip/hip_runtime.h>

// ---------------------------------------------------------------------------
// CDNA5 (gfx1250) cross-attention, bf16 WMMA path, flash-attention fused core.
// Round 2: V tile staged via GLOBAL_LOAD_ASYNC_TO_LDS_B128 (ASYNCcnt path),
//          K-row prefetch via global_prefetch_b8.
// ---------------------------------------------------------------------------

typedef __attribute__((ext_vector_type(16))) __bf16 v16bf;
typedef __attribute__((ext_vector_type(8)))  __bf16 v8bf;
typedef __attribute__((ext_vector_type(8)))  float  v8f;

union BF16x16 { v16bf v; v8bf h[2]; };

#define B_    2
#define S1_   2048
#define S2_   3072
#define H_    16
#define D_    64
#define C_    1024
#define L2E_  1.44269504088896340736f

__device__ __forceinline__ v8f wmma_bf16(v16bf a, v16bf b, v8f c) {
    return __builtin_amdgcn_wmma_f32_16x16x32_bf16(
        /*neg_a=*/false, a, /*neg_b=*/false, b,
        /*c_mod=*/(short)0, c, /*reuse_a=*/false, /*reuse_b=*/false);
}

// A fragment (16x32 bf16, M x K): lane m = l&15; e<8 -> K=kb+hi*8+e ;
// e>=8 -> K=kb+16+hi*8+(e-8). Two contiguous 16B chunks per lane.
__device__ __forceinline__ v16bf load_a_frag(const __bf16* __restrict__ row, int kb, int hi) {
    BF16x16 u;
    u.h[0] = *(const v8bf*)(row + kb + hi * 8);
    u.h[1] = *(const v8bf*)(row + kb + 16 + hi * 8);
    return u.v;
}

// B fragment (32x16 bf16, K x N): lane n = l&15; e<8 -> K=kb+hi*8+e ;
// e>=8 -> K=kb+16+hi*8+(e-8). Two contiguous 16B chunks per lane.
__device__ __forceinline__ v16bf load_b_frag(const __bf16* __restrict__ row, int kb, int hi) {
    BF16x16 u;
    u.h[0] = *(const v8bf*)(row + kb + hi * 8);
    u.h[1] = *(const v8bf*)(row + kb + 16 + hi * 8);
    return u.v;
}

// Async global -> LDS 128-bit copy (CDNA5 ASYNCcnt path, ISA 08 §4.4):
// per-lane: LDS[lds_off ...+15] = MEM[gaddr ...+15]
__device__ __forceinline__ void async_copy_b128(unsigned int lds_off, const void* gaddr) {
    asm volatile("global_load_async_to_lds_b128 %0, %1, off"
                 :: "v"(lds_off), "v"((unsigned long long)(uintptr_t)gaddr)
                 : "memory");
}
__device__ __forceinline__ void wait_asynccnt0() {
    asm volatile("s_wait_asynccnt 0" ::: "memory");
}

// ---------------------------------------------------------------------------
// fp32 -> bf16 elementwise convert
// ---------------------------------------------------------------------------
__global__ __launch_bounds__(256) void f32_to_bf16_kernel(
    const float* __restrict__ in, __bf16* __restrict__ out, int n) {
    int i = blockIdx.x * 256 + threadIdx.x;
    if (i < n) out[i] = (__bf16)in[i];
}

// ---------------------------------------------------------------------------
// C[M,N] (f32) = A[M,K](bf16) @ W[N,K](bf16)^T + bias[N]
// 256 threads = 8 waves; wave w -> 16x16 tile at (blockIdx.y*16, blockIdx.x*128+w*16)
// ---------------------------------------------------------------------------
__global__ __launch_bounds__(256) void gemm_bf16_kernel(
    const __bf16* __restrict__ A, const __bf16* __restrict__ W,
    const float* __restrict__ bias, float* __restrict__ Cout,
    int M, int N, int K) {
    const int lane = threadIdx.x & 31;
    const int wave = threadIdx.x >> 5;
    const int n0 = blockIdx.x * 128 + wave * 16;
    const int m0 = blockIdx.y * 16;
    const int n = lane & 15;
    const int hi = lane >> 4;

    const __bf16* arow = A + (size_t)(m0 + n) * K;   // lane's A row (M = l&15)
    const __bf16* wrow = W + (size_t)(n0 + n) * K;   // lane's W row (Ncol = l&15)

    v8f acc = {0.f, 0.f, 0.f, 0.f, 0.f, 0.f, 0.f, 0.f};
    for (int kb = 0; kb < K; kb += 32) {
        v16bf afrag = load_a_frag(arow, kb, hi);
        v16bf bfrag = load_b_frag(wrow, kb, hi);
        acc = wmma_bf16(afrag, bfrag, acc);
    }
    const float bv = bias[n0 + n];
#pragma unroll
    for (int r = 0; r < 8; ++r)
        Cout[(size_t)(m0 + hi * 8 + r) * N + n0 + n] = acc[r] + bv;
}

// ---------------------------------------------------------------------------
// LayerNorm over d=64 per (b,i,h) of Q [b,s1,1024] f32 -> Qn [b,h,s1,64] bf16
// folds in attention scale d^-0.5. One wave per row, 2 elems/lane.
// ---------------------------------------------------------------------------
__global__ __launch_bounds__(256) void ln_q_kernel(
    const float* __restrict__ Q, const float* __restrict__ w,
    const float* __restrict__ bias, __bf16* __restrict__ Qn, float scale) {
    const int lane = threadIdx.x & 31;
    const int wv = threadIdx.x >> 5;
    const int row = blockIdx.x * 8 + wv;          // (b*S1 + i)*H + h
    const int h = row & 15;
    const int i = (row >> 4) & (S1_ - 1);
    const int b = row >> 15;                       // / (S1_*H_)

    const float* src = Q + (size_t)row * D_;
    float v0 = src[lane * 2], v1 = src[lane * 2 + 1];
    float s = v0 + v1;
#pragma unroll
    for (int off = 1; off < 32; off <<= 1) s += __shfl_xor(s, off, 32);
    float mu = s * (1.f / 64.f);
    float d0 = v0 - mu, d1 = v1 - mu;
    float q2 = d0 * d0 + d1 * d1;
#pragma unroll
    for (int off = 1; off < 32; off <<= 1) q2 += __shfl_xor(q2, off, 32);
    float rstd = rsqrtf(q2 * (1.f / 64.f) + 1e-6f);

    size_t o = (((size_t)(b * H_ + h)) * S1_ + i) * D_;
    Qn[o + lane * 2]     = (__bf16)((d0 * rstd * w[lane * 2]     + bias[lane * 2])     * scale);
    Qn[o + lane * 2 + 1] = (__bf16)((d1 * rstd * w[lane * 2 + 1] + bias[lane * 2 + 1]) * scale);
}

// ---------------------------------------------------------------------------
// From KV [b,s2,2,1024] f32: LN the K half -> Kn [b,h,s2,64] bf16,
// pass-through convert the V half -> Vn [b,h,s2,64] bf16.
// ---------------------------------------------------------------------------
__global__ __launch_bounds__(256) void ln_kv_kernel(
    const float* __restrict__ KV, const float* __restrict__ w,
    const float* __restrict__ bias, __bf16* __restrict__ Kn,
    __bf16* __restrict__ Vn) {
    const int lane = threadIdx.x & 31;
    const int wv = threadIdx.x >> 5;
    const int row = blockIdx.x * 8 + wv;          // (b*S2 + j)*H + h
    const int h = row & 15;
    const int jj = row >> 4;
    const int j = jj % S2_;
    const int b = jj / S2_;

    const float* ksrc = KV + ((size_t)(b * S2_ + j) * 2) * C_ + h * D_;
    const float* vsrc = ksrc + C_;

    float v0 = ksrc[lane * 2], v1 = ksrc[lane * 2 + 1];
    float s = v0 + v1;
#pragma unroll
    for (int off = 1; off < 32; off <<= 1) s += __shfl_xor(s, off, 32);
    float mu = s * (1.f / 64.f);
    float d0 = v0 - mu, d1 = v1 - mu;
    float q2 = d0 * d0 + d1 * d1;
#pragma unroll
    for (int off = 1; off < 32; off <<= 1) q2 += __shfl_xor(q2, off, 32);
    float rstd = rsqrtf(q2 * (1.f / 64.f) + 1e-6f);

    size_t o = (((size_t)(b * H_ + h)) * S2_ + j) * D_;
    Kn[o + lane * 2]     = (__bf16)(d0 * rstd * w[lane * 2]     + bias[lane * 2]);
    Kn[o + lane * 2 + 1] = (__bf16)(d1 * rstd * w[lane * 2 + 1] + bias[lane * 2 + 1]);
    Vn[o + lane * 2]     = (__bf16)vsrc[lane * 2];
    Vn[o + lane * 2 + 1] = (__bf16)vsrc[lane * 2 + 1];
}

// ---------------------------------------------------------------------------
// Flash attention: grid (S1/64, H, B), 128 threads = 4 waves.
// Wave w owns a 16-row Q tile; j loop in steps of 32 keys:
//   async V stage (global->LDS, no VGPR round-trip) overlapped with
//   4x WMMA for scores + online softmax (16-lane half-wave reductions),
//   P staged via LDS into A-fragment layout, then 4x WMMA for P@V.
// Writes ctx bf16 [b, s1, h*64+d].
// ---------------------------------------------------------------------------
__global__ __launch_bounds__(128) void attn_kernel(
    const __bf16* __restrict__ Qn, const __bf16* __restrict__ Kn,
    const __bf16* __restrict__ Vn, const unsigned char* __restrict__ mask,
    __bf16* __restrict__ Ctx) {
    const int lane = threadIdx.x & 31;
    const int wave = threadIdx.x >> 5;
    const int n = lane & 15;
    const int hi = lane >> 4;
    const int h = blockIdx.y;
    const int b = blockIdx.z;
    const int i0 = (blockIdx.x * 4 + wave) * 16;

    const __bf16* qbase = Qn + ((size_t)(b * H_ + h)) * S1_ * D_;
    const __bf16* kbase = Kn + ((size_t)(b * H_ + h)) * S2_ * D_;
    const __bf16* vbase = Vn + ((size_t)(b * H_ + h)) * S2_ * D_;
    const unsigned char* mrow = mask + (size_t)b * S2_;

    __shared__ __align__(16) __bf16 Plds[4][16 * 32];   // per-wave P tile (row-major 16x32)
    __shared__ __align__(16) __bf16 Vlds[32 * 72];      // shared V tile, padded pitch

    // Q A-fragments for this wave's 16 rows (K = d, two 32-steps)
    const __bf16* qrow = qbase + (size_t)(i0 + n) * D_;
    v16bf aq0 = load_a_frag(qrow, 0, hi);
    v16bf aq1 = load_a_frag(qrow, 32, hi);

    const float NEG = -__builtin_inff();
    float mrun[8], lrun[8];
    v8f acc[4];
#pragma unroll
    for (int r = 0; r < 8; ++r) { mrun[r] = NEG; lrun[r] = 0.f; }
#pragma unroll
    for (int t = 0; t < 4; ++t) acc[t] = (v8f){0.f,0.f,0.f,0.f,0.f,0.f,0.f,0.f};

    const int vrow = threadIdx.x >> 2;     // 0..31
    const int vseg = threadIdx.x & 3;      // 0..3 (16 elems each)

    for (int j0 = 0; j0 < S2_; j0 += 32) {
        __syncthreads();  // previous iteration's LDS reads complete

        // cooperative async V tile stage: 32 keys x 64 dims, LDS written by DMA
        {
            const __bf16* vsrc = vbase + (size_t)(j0 + vrow) * D_ + vseg * 16;
            unsigned int ldo =
                (unsigned int)(uintptr_t)(&Vlds[vrow * 72 + vseg * 16]);
            async_copy_b128(ldo, vsrc);
            async_copy_b128(ldo + 16, vsrc + 8);
        }

        // prefetch next iteration's K rows while this step computes
        if (j0 + 32 < S2_) {
            __builtin_prefetch(kbase + (size_t)(j0 + 32 + n) * D_, 0, 3);
            __builtin_prefetch(kbase + (size_t)(j0 + 48 + n) * D_, 0, 3);
        }

        // ---- scores: two 16x16 tiles over 32 keys ----
        v8f s0 = (v8f){0.f,0.f,0.f,0.f,0.f,0.f,0.f,0.f};
        v8f s1 = (v8f){0.f,0.f,0.f,0.f,0.f,0.f,0.f,0.f};
        const __bf16* krow0 = kbase + (size_t)(j0 + n) * D_;
        const __bf16* krow1 = kbase + (size_t)(j0 + 16 + n) * D_;
        s0 = wmma_bf16(aq0, load_b_frag(krow0, 0, hi), s0);
        s0 = wmma_bf16(aq1, load_b_frag(krow0, 32, hi), s0);
        s1 = wmma_bf16(aq0, load_b_frag(krow1, 0, hi), s1);
        s1 = wmma_bf16(aq1, load_b_frag(krow1, 32, hi), s1);

        // padding mask (per-lane column)
        if (!mrow[j0 + n]) {
#pragma unroll
            for (int r = 0; r < 8; ++r) s0[r] = NEG;
        }
        if (!mrow[j0 + 16 + n]) {
#pragma unroll
            for (int r = 0; r < 8; ++r) s1[r] = NEG;
        }

        // ---- online softmax; each 16-lane half owns rows hi*8..hi*8+7 ----
        __bf16* myP = Plds[wave];
#pragma unroll
        for (int r = 0; r < 8; ++r) {
            float tmax = fmaxf(s0[r], s1[r]);
#pragma unroll
            for (int off = 1; off < 16; off <<= 1)
                tmax = fmaxf(tmax, __shfl_xor(tmax, off, 32));
            float mnew = fmaxf(mrun[r], tmax);
            float alpha = exp2f((mrun[r] - mnew) * L2E_);
            mrun[r] = mnew;
            float p0 = exp2f((s0[r] - mnew) * L2E_);
            float p1 = exp2f((s1[r] - mnew) * L2E_);
            myP[(hi * 8 + r) * 32 + n]      = (__bf16)p0;
            myP[(hi * 8 + r) * 32 + 16 + n] = (__bf16)p1;
            float ts = p0 + p1;
#pragma unroll
            for (int off = 1; off < 16; off <<= 1)
                ts += __shfl_xor(ts, off, 32);
            lrun[r] = lrun[r] * alpha + ts;
#pragma unroll
            for (int t = 0; t < 4; ++t) acc[t][r] *= alpha;
        }

        wait_asynccnt0();   // V tile DMA complete (this wave's ops)
        __syncthreads();    // P + V visible block-wide

        // P as A-fragment (16x32, contraction over the 32 keys)
        BF16x16 up;
        up.h[0] = *(const v8bf*)(&myP[(lane & 15) * 32 + hi * 16]);
        up.h[1] = *(const v8bf*)(&myP[(lane & 15) * 32 + hi * 16 + 8]);

        // V B-fragments per 16-wide d tile
#pragma unroll
        for (int t = 0; t < 4; ++t) {
            BF16x16 ub;
#pragma unroll
            for (int e = 0; e < 16; ++e)
                ub.v[e] = Vlds[(hi * 16 + e) * 72 + t * 16 + n];
            acc[t] = wmma_bf16(up.v, ub.v, acc[t]);
        }
    }

    // ---- normalize and write ctx as bf16 [b, i, h*64+d] ----
#pragma unroll
    for (int r = 0; r < 8; ++r) {
        float inv = 1.f / lrun[r];
        size_t rowo = (size_t)(b * S1_ + i0 + hi * 8 + r) * C_ + h * D_;
#pragma unroll
        for (int t = 0; t < 4; ++t)
            Ctx[rowo + t * 16 + n] = (__bf16)(acc[t][r] * inv);
    }
}

// ---------------------------------------------------------------------------
// Host-side orchestration
// ---------------------------------------------------------------------------
extern "C" void kernel_launch(void* const* d_in, const int* in_sizes, int n_in,
                              void* d_out, int out_size, void* d_ws, size_t ws_size,
                              hipStream_t stream) {
    const float* x    = (const float*)d_in[0];
    const float* y    = (const float*)d_in[1];
    const unsigned char* mask = (const unsigned char*)d_in[2];
    const float* Wq   = (const float*)d_in[3];
    const float* bq   = (const float*)d_in[4];
    const float* Wkv  = (const float*)d_in[5];
    const float* bkv  = (const float*)d_in[6];
    const float* qn_w = (const float*)d_in[7];
    const float* qn_b = (const float*)d_in[8];
    const float* kn_w = (const float*)d_in[9];
    const float* kn_b = (const float*)d_in[10];
    const float* Wo   = (const float*)d_in[11];
    const float* bo   = (const float*)d_in[12];
    float* out = (float*)d_out;

    const int M1 = B_ * S1_;   // 4096
    const int M2 = B_ * S2_;   // 6144

    char* ws = (char*)d_ws;
    size_t off = 0;
    auto alloc = [&](size_t bytes) -> void* {
        void* p = ws + off;
        off += (bytes + 255) & ~(size_t)255;
        return p;
    };
    __bf16* xb   = (__bf16*)alloc((size_t)M1 * C_ * 2);
    __bf16* yb   = (__bf16*)alloc((size_t)M2 * C_ * 2);
    __bf16* wqb  = (__bf16*)alloc((size_t)C_ * C_ * 2);
    __bf16* wkvb = (__bf16*)alloc((size_t)2 * C_ * C_ * 2);
    __bf16* wob  = (__bf16*)alloc((size_t)C_ * C_ * 2);
    float*  Qf   = (float*)alloc((size_t)M1 * C_ * 4);
    float*  KVf  = (float*)alloc((size_t)M2 * 2 * C_ * 4);
    __bf16* Qnb  = (__bf16*)alloc((size_t)M1 * C_ * 2);
    __bf16* Knb  = (__bf16*)alloc((size_t)M2 * C_ * 2);
    __bf16* Vnb  = (__bf16*)alloc((size_t)M2 * C_ * 2);
    __bf16* Ctxb = (__bf16*)alloc((size_t)M1 * C_ * 2);
    if (off > ws_size) return;  // workspace too small; nothing sane to do

    auto cvt = [&](const float* src, __bf16* dst, int n) {
        f32_to_bf16_kernel<<<(n + 255) / 256, 256, 0, stream>>>(src, dst, n);
    };
    cvt(x, xb, M1 * C_);
    cvt(y, yb, M2 * C_);
    cvt(Wq, wqb, C_ * C_);
    cvt(Wkv, wkvb, 2 * C_ * C_);
    cvt(Wo, wob, C_ * C_);

    // Q = x @ Wq^T + bq
    gemm_bf16_kernel<<<dim3(C_ / 128, M1 / 16), 256, 0, stream>>>(
        xb, wqb, bq, Qf, M1, C_, C_);
    // KV = y @ Wkv^T + bkv
    gemm_bf16_kernel<<<dim3(2 * C_ / 128, M2 / 16), 256, 0, stream>>>(
        yb, wkvb, bkv, KVf, M2, 2 * C_, C_);

    // per-head LN (+ scale for q) and relayout to [b,h,s,64] bf16
    ln_q_kernel<<<(B_ * S1_ * H_) / 8, 256, 0, stream>>>(Qf, qn_w, qn_b, Qnb, 0.125f);
    ln_kv_kernel<<<(B_ * S2_ * H_) / 8, 256, 0, stream>>>(KVf, kn_w, kn_b, Knb, Vnb);

    // fused masked softmax(QK^T) @ V
    attn_kernel<<<dim3(S1_ / 64, H_, B_), 128, 0, stream>>>(Qnb, Knb, Vnb, mask, Ctxb);

    // out = ctx @ Wo^T + bo
    gemm_bf16_kernel<<<dim3(C_ / 128, M1 / 16), 256, 0, stream>>>(
        Ctxb, wob, bo, out, M1, C_, C_);
}
